// MultiHeadAttention_49091476193504
// MI455X (gfx1250) — matrix operand order repaired and checked by reading, not compiled, not run on hardware
//
#include <hip/hip_runtime.h>
#include <hip/hip_bf16.h>
#include <math.h>
#include <stdint.h>

typedef __attribute__((ext_vector_type(16))) _Float16 v16h;
typedef __attribute__((ext_vector_type(8)))  _Float16 v8h;
typedef __attribute__((ext_vector_type(8)))  float    v8f;
typedef int b128_t __attribute__((vector_size(16)));   // matches builtin param

#define B_    2
#define S_    2048
#define D_    1024
#define H_    16
#define DH_   64
#define MTOT  (B_ * S_)        // 4096 rows of x / ctx

union AFrag {
    v16h v;
    v8h  h2[2];
    _Float16 e[16];
};

__device__ __forceinline__ v8f wmma_f16(v16h a, v16h b, v8f c) {
    // D = A(16x32 f16) x B(32x16 f16) + C(16x16 f32)
    return __builtin_amdgcn_wmma_f32_16x16x32_f16(
        /*neg_a=*/false, a, /*neg_b=*/false, b,
        /*c_mod=*/(short)0, c, /*reuse_a=*/false, /*reuse_b=*/false);
}

// 16-byte global -> LDS copy, using the CDNA5 async path when available.
__device__ __forceinline__ void lds_copy16(_Float16* dst, const _Float16* src) {
#if defined(__AMDGCN__) && __has_builtin(__builtin_amdgcn_global_load_async_to_lds_b128)
    __builtin_amdgcn_global_load_async_to_lds_b128(
        (__attribute__((address_space(1))) b128_t*)(uintptr_t)src,
        (__attribute__((address_space(3))) b128_t*)(uintptr_t)dst,
        /*offset=*/0, /*cpol=*/0);
#else
    *(v8h*)dst = *(const v8h*)src;
#endif
}

// Wait until at most N async copies remain outstanding (in-order completion).
template <int N>
__device__ __forceinline__ void async_wait_le() {
#if defined(__AMDGCN__) && __has_builtin(__builtin_amdgcn_global_load_async_to_lds_b128)
#if __has_builtin(__builtin_amdgcn_s_wait_asynccnt)
    __builtin_amdgcn_s_wait_asynccnt(N);
#else
    asm volatile("s_wait_asynccnt %0" ::"i"(N) : "memory");
#endif
#endif
}

// ---------------------------------------------------------------------------
// fp32 -> fp16 conversion (x and weights)
// ---------------------------------------------------------------------------
__global__ void cvt_f32_f16(const float* __restrict__ src,
                            _Float16* __restrict__ dst, int n) {
    for (int i = blockIdx.x * blockDim.x + threadIdx.x; i < n;
         i += gridDim.x * blockDim.x)
        dst[i] = (_Float16)src[i];
}

// ---------------------------------------------------------------------------
// WMMA GEMM:  C[M,N] = A[M,K] * W[N,K]^T + bias   (torch Linear layout)
// MODE 0: store f16, head-major       [B,H,S,DH]   (Q, K projections)
// MODE 1: store f32, row-major        [M,N]        (output projection)
// MODE 2: store f16, head-TRANSPOSED  [B,H,DH,S]   (V projection -> V^T)
// Block: 256 thr = 8 waves (4 M x 2 N), block tile 128x128, wave tile 32x64.
// ---------------------------------------------------------------------------
template <int MODE>
__global__ __launch_bounds__(256)
void gemm_ws(const _Float16* __restrict__ A,
             const _Float16* __restrict__ W,
             const float* __restrict__ bias,
             void* __restrict__ out,
             int M, int N, int K) {
    const int lane = threadIdx.x & 31;
    const int wid  = threadIdx.x >> 5;
    const int l15  = lane & 15;
    const int hi   = lane >> 4;           // lane half: K-range selector
    const int wm   = wid & 3;
    const int wn   = wid >> 2;
    const int m0   = blockIdx.y * 128 + wm * 32;
    const int n0   = blockIdx.x * 128 + wn * 64;

    const v8f zero = {};
    v8f c[2][4];
#pragma unroll
    for (int i = 0; i < 2; ++i)
#pragma unroll
        for (int j = 0; j < 4; ++j) c[i][j] = zero;

    for (int k0 = 0; k0 < K; k0 += 32) {
        AFrag a[2];
#pragma unroll
        for (int mt = 0; mt < 2; ++mt) {
            const _Float16* ap = A + (size_t)(m0 + mt * 16 + l15) * K + k0;
            // A layout: lo lanes K 0..15, hi lanes K 16..31
            a[mt].h2[0] = *(const v8h*)(ap + hi * 16);
            a[mt].h2[1] = *(const v8h*)(ap + hi * 16 + 8);
        }
        v16h b[4];
#pragma unroll
        for (int nt = 0; nt < 4; ++nt) {
            // B layout: lane holds column n, halves h -> K = hi*16 + h
            const _Float16* bp =
                W + (size_t)(n0 + nt * 16 + l15) * K + k0 + hi * 16;
            b[nt] = *(const v16h*)bp;
        }
        if (k0 + 32 < K) {
            __builtin_prefetch(A + (size_t)(m0 + l15) * K + k0 + 32, 0, 0);
            __builtin_prefetch(W + (size_t)(n0 + l15) * K + k0 + 32, 0, 0);
        }
#pragma unroll
        for (int mt = 0; mt < 2; ++mt)
#pragma unroll
            for (int nt = 0; nt < 4; ++nt)
                c[mt][nt] = wmma_f16(a[mt].v, b[nt], c[mt][nt]);
    }

    // Epilogue: C layout — lane holds column n = l15, rows m = r + 8*hi
#pragma unroll
    for (int mt = 0; mt < 2; ++mt)
#pragma unroll
        for (int nt = 0; nt < 4; ++nt) {
            const int n = n0 + nt * 16 + l15;
            const float bv = bias[n];
#pragma unroll
            for (int r = 0; r < 8; ++r) {
                const int m = m0 + mt * 16 + r + 8 * hi;
                const float val = c[mt][nt][r] + bv;
                const int b  = m >> 11;       // m / S_
                const int s  = m & (S_ - 1);
                const int h  = n >> 6;        // n / DH_
                const int dh = n & (DH_ - 1);
                if (MODE == 0) {
                    ((_Float16*)out)[(((size_t)b * H_ + h) * S_ + s) * DH_ +
                                     dh] = (_Float16)val;
                } else if (MODE == 2) {
                    ((_Float16*)out)[(((size_t)b * H_ + h) * DH_ + dh) * S_ +
                                     s] = (_Float16)val;
                } else {
                    ((float*)out)[(size_t)m * N + n] = val;
                }
            }
        }
}

// ---------------------------------------------------------------------------
// Flash attention: Q/K f16 [B,H,S,64], V^T f16 [B,H,64,S], scale = 1/8.
// Block = 128 thr (4 waves) on one (b,h); wave owns a 16-row q tile.
// K and V^T chunks streamed through DOUBLE-BUFFERED async global->LDS copies:
// chunk i+1 is issued, then s_wait_asynccnt<=4 guarantees chunk i landed
// (async loads complete in order), so WMMA on chunk i overlaps the stream-in.
// Energy computed TRANSPOSED (E^T = K_tile x Q^T) so its C-layout equals the
// A-layout of P for the P*V WMMA -> probs convert per-lane, no transpose.
// Output ctx f16 in [B,S,H*DH] (row-major, GEMM-ready).
// ---------------------------------------------------------------------------
__global__ __launch_bounds__(128)
void flash_attn(const _Float16* __restrict__ Qg,
                const _Float16* __restrict__ Kg,
                const _Float16* __restrict__ Vtg,
                _Float16* __restrict__ ctx) {
    __shared__ _Float16 sK[2][32][80];    // K chunk  [s_k=32][dh=64]
    __shared__ _Float16 sVt[2][64][48];   // V^T chunk [dh=64][s_k=32]

    const int lane = threadIdx.x & 31;
    const int wid  = threadIdx.x >> 5;     // 0..3
    const int l15  = lane & 15;
    const int hi   = lane >> 4;

    const int bh  = blockIdx.x >> 5;       // 0..31  (b*16 + h)
    const int qc  = blockIdx.x & 31;       // q chunk of 64 rows
    const int b   = bh >> 4;
    const int h   = bh & 15;
    const int sq0 = qc * 64 + wid * 16;    // this wave's 16 query rows

    const _Float16* Qb  = Qg  + (size_t)bh * S_ * DH_;
    const _Float16* Kb  = Kg  + (size_t)bh * S_ * DH_;
    const _Float16* Vtb = Vtg + (size_t)bh * DH_ * S_;

    // Q as WMMA B-operand (in registers for the whole loop):
    // lane -> column s_q = sq0 + l15; halves -> dh = dh0 + hi*16 + h
    v16h qB[2];
    {
        const _Float16* qp = Qb + (size_t)(sq0 + l15) * DH_;
        qB[0] = *(const v16h*)(qp + 0  + hi * 16);
        qB[1] = *(const v16h*)(qp + 32 + hi * 16);
    }

    const v8f zero = {};
    v8f o[4];
#pragma unroll
    for (int t = 0; t < 4; ++t) o[t] = zero;
    float m_run = -INFINITY;
    float l_run = 0.0f;

    const int krow = threadIdx.x >> 2;          // 0..31
    const int kcol = (threadIdx.x & 3) * 16;    // 0,16,32,48
    const int vrow = threadIdx.x >> 1;          // 0..63
    const int vcol = (threadIdx.x & 1) * 16;    // 0,16

    // 4 async copies per thread per chunk
    auto stage = [&](int kc, int buf) {
        const _Float16* gk = Kb + (size_t)(kc + krow) * DH_ + kcol;
        const _Float16* gv = Vtb + (size_t)vrow * S_ + kc + vcol;
        lds_copy16(&sK[buf][krow][kcol],     gk);
        lds_copy16(&sK[buf][krow][kcol + 8], gk + 8);
        lds_copy16(&sVt[buf][vrow][vcol],     gv);
        lds_copy16(&sVt[buf][vrow][vcol + 8], gv + 8);
    };

    stage(0, 0);   // prologue: chunk 0 -> buffer 0

    for (int kc = 0, it = 0; kc < S_; kc += 32, ++it) {
        const int cur = it & 1;
        const int nxt = cur ^ 1;
        // issue next chunk into the other buffer (safe: last read of that
        // buffer was before the trailing barrier of the previous iteration)
        const int kcn = (kc + 32 < S_) ? kc + 32 : kc;
        stage(kcn, nxt);
        if (kc + 64 < S_) {
            __builtin_prefetch(Kb + (size_t)(kc + 64 + krow) * DH_ + kcol, 0, 0);
            __builtin_prefetch(Vtb + (size_t)vrow * S_ + kc + 64 + vcol, 0, 0);
        }
        async_wait_le<4>();   // the 4 newest are chunk i+1; chunk i is done
        __syncthreads();

        // ---- E^T tiles: tile tt covers s_k = kc + tt*16 .. +16
        AFrag ka[2][2];
#pragma unroll
        for (int tt = 0; tt < 2; ++tt) {
            const _Float16* kp = &sK[cur][tt * 16 + l15][0];
            ka[tt][0].h2[0] = *(const v8h*)(kp + hi * 16);         // dh 0..31
            ka[tt][0].h2[1] = *(const v8h*)(kp + hi * 16 + 8);
            ka[tt][1].h2[0] = *(const v8h*)(kp + 32 + hi * 16);    // dh 32..63
            ka[tt][1].h2[1] = *(const v8h*)(kp + 40 + hi * 16);
        }
        v8f et0 = zero, et1 = zero;
        et0 = wmma_f16(ka[0][0].v, qB[0], et0);
        et0 = wmma_f16(ka[0][1].v, qB[1], et0);
        et1 = wmma_f16(ka[1][0].v, qB[0], et1);
        et1 = wmma_f16(ka[1][1].v, qB[1], et1);

        // ---- online softmax (per lane: column s_q = l15; 8+8 s_k rows)
        const float sm_scale = 0.125f;   // 1/sqrt(64)
#pragma unroll
        for (int r = 0; r < 8; ++r) {
            et0[r] = et0[r] * sm_scale;
            et1[r] = et1[r] * sm_scale;
        }
        float mloc = -INFINITY;
#pragma unroll
        for (int r = 0; r < 8; ++r) {
            mloc = fmaxf(mloc, et0[r]);
            mloc = fmaxf(mloc, et1[r]);
        }
        mloc = fmaxf(mloc, __shfl_xor(mloc, 16, 32));  // both row halves
        const float m_new = fmaxf(m_run, mloc);
        const float alpha = __expf(m_run - m_new);

        AFrag pA;   // P in A-layout: halves 0..7 <- et0 regs, 8..15 <- et1
        float ssum = 0.0f;
#pragma unroll
        for (int r = 0; r < 8; ++r) {
            const float p0 = __expf(et0[r] - m_new);
            const float p1 = __expf(et1[r] - m_new);
            ssum += p0 + p1;
            pA.e[r]     = (_Float16)p0;
            pA.e[8 + r] = (_Float16)p1;
        }
        ssum += __shfl_xor(ssum, 16, 32);
        l_run = l_run * alpha + ssum;
        m_run = m_new;

        // rescale O accumulator: row r+8*hi needs alpha of lane (r+8*hi)
        float al[8];
#pragma unroll
        for (int r = 0; r < 8; ++r) al[r] = __shfl(alpha, r + 8 * hi, 32);
#pragma unroll
        for (int t = 0; t < 4; ++t)
#pragma unroll
            for (int r = 0; r < 8; ++r) o[t][r] *= al[r];

        // ---- P * V: B-frag = two v8h pieces of a V^T row in LDS
        //      lane -> column dh = t*16 + l15; s_k = hi*8 + h (h<8), 16 + hi*8 + (h-8)
#pragma unroll
        for (int t = 0; t < 4; ++t) {
            AFrag vb;
            vb.h2[0] = *(const v8h*)&sVt[cur][t * 16 + l15][hi * 8];
            vb.h2[1] = *(const v8h*)&sVt[cur][t * 16 + l15][16 + hi * 8];
            o[t] = wmma_f16(pA.v, vb.v, o[t]);
        }
        __syncthreads();   // everyone done reading buf 'cur' before reuse
    }

    // ---- normalize and store ctx[b, s, h*64 + dh] as f16
    const float rl = 1.0f / l_run;
    float li[8];
#pragma unroll
    for (int r = 0; r < 8; ++r) li[r] = __shfl(rl, r + 8 * hi, 32);
#pragma unroll
    for (int t = 0; t < 4; ++t)
#pragma unroll
        for (int r = 0; r < 8; ++r) {
            const int s = sq0 + r + 8 * hi;
            ctx[((size_t)b * S_ + s) * D_ + h * DH_ + t * 16 + l15] =
                (_Float16)(o[t][r] * li[r]);
        }
}

// ---------------------------------------------------------------------------
extern "C" void kernel_launch(void* const* d_in, const int* in_sizes, int n_in,
                              void* d_out, int out_size, void* d_ws,
                              size_t ws_size, hipStream_t stream) {
    (void)in_sizes; (void)n_in; (void)out_size; (void)ws_size;
    const float* x  = (const float*)d_in[0];
    const float* Wq = (const float*)d_in[1];
    const float* bq = (const float*)d_in[2];
    const float* Wk = (const float*)d_in[3];
    const float* bk = (const float*)d_in[4];
    const float* Wv = (const float*)d_in[5];
    const float* bv = (const float*)d_in[6];
    const float* Wo = (const float*)d_in[7];
    const float* bo = (const float*)d_in[8];

    char* ws = (char*)d_ws;
    const size_t MB = 1024 * 1024;
    _Float16* xh  = (_Float16*)(ws + 0);        // 8 MiB  [4096,1024]
    _Float16* wqh = (_Float16*)(ws + 8 * MB);   // 2 MiB  [1024,1024]
    _Float16* wkh = (_Float16*)(ws + 10 * MB);
    _Float16* wvh = (_Float16*)(ws + 12 * MB);
    _Float16* woh = (_Float16*)(ws + 14 * MB);
    _Float16* Qh  = (_Float16*)(ws + 16 * MB);  // 8 MiB  [B,H,S,64]
    _Float16* Kh  = (_Float16*)(ws + 24 * MB);  // 8 MiB  [B,H,S,64]
    _Float16* Vth = (_Float16*)(ws + 32 * MB);  // 8 MiB  [B,H,64,S]
    _Float16* ctx = (_Float16*)(ws + 40 * MB);  // 8 MiB  [4096,1024]

    cvt_f32_f16<<<1024, 256, 0, stream>>>(x,  xh,  MTOT * D_);
    cvt_f32_f16<<<512,  256, 0, stream>>>(Wq, wqh, D_ * D_);
    cvt_f32_f16<<<512,  256, 0, stream>>>(Wk, wkh, D_ * D_);
    cvt_f32_f16<<<512,  256, 0, stream>>>(Wv, wvh, D_ * D_);
    cvt_f32_f16<<<512,  256, 0, stream>>>(Wo, woh, D_ * D_);

    dim3 gg(D_ / 128, MTOT / 128);  // (8, 32)
    gemm_ws<0><<<gg, 256, 0, stream>>>(xh, wqh, bq, Qh,  MTOT, D_, D_);
    gemm_ws<0><<<gg, 256, 0, stream>>>(xh, wkh, bk, Kh,  MTOT, D_, D_);
    gemm_ws<2><<<gg, 256, 0, stream>>>(xh, wvh, bv, Vth, MTOT, D_, D_);

    flash_attn<<<B_ * H_ * (S_ / 64), 128, 0, stream>>>(Qh, Kh, Vth, ctx);

    gemm_ws<1><<<gg, 256, 0, stream>>>(ctx, woh, bo, d_out, MTOT, D_, D_);
}
